// ProcessHandBoard_24704651887294
// MI455X (gfx1250) — compile-verified, matrix operand order checked by reading, not run on hardware
//
#include <hip/hip_runtime.h>

typedef _Float16 f16;
typedef f16  v8h  __attribute__((ext_vector_type(8)));
typedef f16  v16h __attribute__((ext_vector_type(16)));
typedef float v8f __attribute__((ext_vector_type(8)));
typedef unsigned int v4u __attribute__((ext_vector_type(4)));
typedef int  v8i_ __attribute__((ext_vector_type(8)));
typedef int  v4i_ __attribute__((ext_vector_type(4)));

#define N_HERO 7680            // B*M = 128*60
#define N_ROWS (2 * N_HERO)    // hero rows then villain rows
#define K1     2048            // feature width (128 ch * 16)
#define NF     512             // hidden width
#define EPSBN  1e-5f

// LDS A-tile geometry: 32 rows x 32 f16, TDM-padded to 80B row stride
// (pad_interval = 16 dwords (one 64B row), pad_amount = 4 dwords (16B)).
#define TILE_ROW_F16 40        // 80 bytes / 2
#define TILE_BYTES   2560      // 32 * 80

// Load one lane's share of a 16x32 f16 WMMA operand from a k-contiguous row.
// Lane half 0 holds K = {0..7, 16..23}, half 1 holds K = {8..15, 24..31}.
__device__ __forceinline__ v16h load_frag(const f16* p, int half) {
    union { v16h v; v8h h[2]; } f;
    f.h[0] = *(const v8h*)(p + half * 8);
    f.h[1] = *(const v8h*)(p + half * 8 + 16);
    return f.v;
}

#define WMMA_F16(a, b, c) \
    __builtin_amdgcn_wmma_f32_16x16x32_f16(false, (a), false, (b), (short)0, (c), false, false)

// ---------------------------------------------------------------------------
// TDM: 2D tensor_load_to_lds of a 32x32 f16 tile (rows K1-strided in global)
// into LDS at byte offset `ldsoff`, padding 16B after every 64B row so later
// ds_load_b128 row reads are bank-conflict-free. D# per CDNA5 ISA 8.3/8.4.
// ---------------------------------------------------------------------------
__device__ __forceinline__ void tdm_load_a_tile(unsigned long long gaddr,
                                                unsigned ldsoff) {
    v4u g0;
    g0.x = 1u;                                   // count=1, user descriptor
    g0.y = ldsoff;                               // lds_addr (bytes)
    g0.z = (unsigned)gaddr;                      // global_addr[31:0]
    g0.w = (unsigned)(gaddr >> 32) | (2u << 30); // global_addr[56:32] | type=2

    v8i_ g1;
    g1[0] = (int)((1u << 16)        // data_size = 1 -> 2 bytes
                | (1u << 20)        // pad_enable
                | (3u << 22)        // pad_interval: code 3 -> 16 dwords (64B)
                | (3u << 25));      // pad_amount:   code 3 -> 4 dwords (16B)
    g1[1] = (int)((K1 & 0xFFFF) << 16);          // tensor_dim0[15:0]
    g1[2] = (int)((K1 >> 16) | (32u << 16));     // td0[31:16] | tensor_dim1[15:0]
    g1[3] = (int)(32u << 16);                    // td1[31:16]=0 | tile_dim0=32
    g1[4] = 32;                                  // tile_dim1=32, tile_dim2=0
    g1[5] = K1;                                  // tensor_dim0_stride[31:0]
    g1[6] = 0;                                   // stride0[47:32] | stride1 lo
    g1[7] = 0;
    v4i_ z4 = {0, 0, 0, 0};                      // groups 2/3 unused (2D)
    v8i_ z8 = {0, 0, 0, 0, 0, 0, 0, 0};
    __builtin_amdgcn_tensor_load_to_lds(g0, g1, z4, z4, z8, 0);
}

// ---------------------------------------------------------------------------
// Kernel 1: transpose + convert w1 (2048x512) and w2 (512x512) to f16 Wt[n][k]
// ---------------------------------------------------------------------------
__global__ void prep_weights(const float* __restrict__ w1, const float* __restrict__ w2,
                             f16* __restrict__ Wt1, f16* __restrict__ Wt2) {
    int i = blockIdx.x * blockDim.x + threadIdx.x;     // 1,310,720 threads exactly
    if (i < K1 * NF) {
        int n = i / K1, k = i % K1;
        Wt1[i] = (f16)w1[(size_t)k * NF + n];
    } else {
        int j = i - K1 * NF;                           // < NF*NF
        int n = j / NF, k = j % NF;
        Wt2[j] = (f16)w2[(size_t)k * NF + n];
    }
}

// ---------------------------------------------------------------------------
// Kernel 2: build H (N_ROWS x 2048, f16). One thread per (row, channel).
// Conv-over-one-hot == lookups; BN folded to per-channel affine; ReLU fused.
// ---------------------------------------------------------------------------
__global__ void features(const int* __restrict__ x,
                         const float* __restrict__ ws_w, const float* __restrict__ ws_b,
                         const float* __restrict__ wr_w, const float* __restrict__ wr_b,
                         const float* __restrict__ sg, const float* __restrict__ sb,
                         const float* __restrict__ sm, const float* __restrict__ sv,
                         const float* __restrict__ rg, const float* __restrict__ rb,
                         const float* __restrict__ rm, const float* __restrict__ rv,
                         f16* __restrict__ H) {
    int gid = blockIdx.x * blockDim.x + threadIdx.x;
    int n = gid >> 7;          // combined row: 0..15359
    int o = gid & 127;         // channel
    int base = (n >= N_HERO) ? (n - N_HERO) : n;   // sample index b*60+m
    int roff = (n >= N_HERO) ? 4 : 0;              // villain uses cards 2,3
    const int* xp = x + base * 14;

    int ridx[5], sidx[5];
    ridx[0] = xp[roff];     ridx[1] = xp[roff + 2];
    ridx[2] = xp[8];        ridx[3] = xp[10];       ridx[4] = xp[12];
    sidx[0] = xp[roff + 1]; sidx[1] = xp[roff + 3];
    sidx[2] = xp[9];        sidx[3] = xp[11];       sidx[4] = xp[13];

    float wr[25];
#pragma unroll
    for (int t = 0; t < 25; ++t) wr[t] = wr_w[o * 25 + t];

    float inv_r = rg[o] / sqrtf(rv[o] + EPSBN);
    float add_r = (wr_b[o] - rm[o]) * inv_r + rb[o];
    float inv_s = sg[o] / sqrtf(sv[o] + EPSBN);
    float add_s = (ws_b[o] - sm[o]) * inv_s + sb[o];

    f16 outv[16] __attribute__((aligned(16)));
#pragma unroll
    for (int p = 0; p < 11; ++p) {                 // rank conv, valid positions
        float acc = 0.f;
#pragma unroll
        for (int c = 0; c < 5; ++c) {
            int d = ridx[c] - p;
            if (d >= 0 && d < 5) acc += wr[c * 5 + d];
        }
        outv[p] = (f16)fmaxf(acc * inv_r + add_r, 0.f);
    }
#pragma unroll
    for (int l = 0; l < 5; ++l) {                  // suit "conv" (k=1)
        outv[11 + l] = (f16)fmaxf(ws_w[o * 5 + sidx[l]] * inv_s + add_s, 0.f);
    }
    v8h* dst = (v8h*)(H + (size_t)n * K1 + o * 16);
    dst[0] = *(const v8h*)&outv[0];
    dst[1] = *(const v8h*)&outv[8];
}

// ---------------------------------------------------------------------------
// Kernel 3: GEMM1 (N_ROWS x 2048) @ (2048 x 512) + bias + ReLU -> f16 A1.
// Wave computes 32(M) x 64(N); A tiles staged by TDM into double-buffered LDS
// (per-wave buffers, s_wait_tensorcnt pipelined, 2-chunk unroll so each body
// uses a fixed buffer), B frags straight from L2 as one clause per chunk.
// ---------------------------------------------------------------------------
__device__ __forceinline__ void gemm1_chunk(const f16* at, const f16* bt,
                                            int half, int l16,
                                            v8f& c00, v8f& c01, v8f& c02, v8f& c03,
                                            v8f& c10, v8f& c11, v8f& c12, v8f& c13) {
    v16h a0 = load_frag(at + l16 * TILE_ROW_F16, half);
    v16h a1 = load_frag(at + (16 + l16) * TILE_ROW_F16, half);
    v16h b0 = load_frag(bt, half);
    v16h b1 = load_frag(bt + (size_t)16 * K1, half);
    v16h b2 = load_frag(bt + (size_t)32 * K1, half);
    v16h b3 = load_frag(bt + (size_t)48 * K1, half);
    c00 = WMMA_F16(a0, b0, c00);
    c10 = WMMA_F16(a1, b0, c10);
    c01 = WMMA_F16(a0, b1, c01);
    c11 = WMMA_F16(a1, b1, c11);
    c02 = WMMA_F16(a0, b2, c02);
    c12 = WMMA_F16(a1, b2, c12);
    c03 = WMMA_F16(a0, b3, c03);
    c13 = WMMA_F16(a1, b3, c13);
}

__global__ void __launch_bounds__(256, 1) gemm1(const f16* __restrict__ A,
                                                const f16* __restrict__ Bt,
                                                const float* __restrict__ bias,
                                                f16* __restrict__ Out) {
    __shared__ f16 abuf[8][2][TILE_BYTES / 2];     // 40 KB: 8 waves x 2 bufs

    const int lane = threadIdx.x & 31;
    const int wave = threadIdx.x >> 5;
    const int half = lane >> 4;
    const int l16  = lane & 15;
    const int row0 = blockIdx.x * 32;              // 480 blocks cover 15360 rows
    const int n0   = wave * 64;

    const f16* bp = Bt + (size_t)(n0 + l16) * K1;

    unsigned lds0 = (unsigned)__builtin_amdgcn_readfirstlane(
        (int)(unsigned long long)&abuf[wave][0][0]);
    unsigned lds1 = lds0 + TILE_BYTES;
    unsigned long long ga =
        (unsigned long long)(const void*)(A + (size_t)row0 * K1);
    const f16* at0 = &abuf[wave][0][0];
    const f16* at1 = &abuf[wave][1][0];

    v8f c00 = {}, c01 = {}, c02 = {}, c03 = {};
    v8f c10 = {}, c11 = {}, c12 = {}, c13 = {};

    tdm_load_a_tile(ga, lds0);                     // prefetch chunk 0
    tdm_load_a_tile(ga + 64, lds1);                // prefetch chunk 1
    for (int t = 0; t < 62; t += 2) {
        __builtin_amdgcn_s_wait_tensorcnt((short)1);   // chunk t landed
        asm volatile("" ::: "memory");
        gemm1_chunk(at0, bp + t * 32, half, l16,
                    c00, c01, c02, c03, c10, c11, c12, c13);
        tdm_load_a_tile(ga + (unsigned long long)(t + 2) * 64, lds0);
        __builtin_amdgcn_s_wait_tensorcnt((short)1);   // chunk t+1 landed
        asm volatile("" ::: "memory");
        gemm1_chunk(at1, bp + (t + 1) * 32, half, l16,
                    c00, c01, c02, c03, c10, c11, c12, c13);
        tdm_load_a_tile(ga + (unsigned long long)(t + 3) * 64, lds1);
    }
    __builtin_amdgcn_s_wait_tensorcnt((short)1);       // chunk 62 landed
    asm volatile("" ::: "memory");
    gemm1_chunk(at0, bp + 62 * 32, half, l16,
                c00, c01, c02, c03, c10, c11, c12, c13);
    __builtin_amdgcn_s_wait_tensorcnt((short)0);       // chunk 63 landed
    asm volatile("" ::: "memory");
    gemm1_chunk(at1, bp + 63 * 32, half, l16,
                c00, c01, c02, c03, c10, c11, c12, c13);

    {
        const v8f cr0[4] = {c00, c01, c02, c03};
        const v8f cr1[4] = {c10, c11, c12, c13};
#pragma unroll
        for (int j = 0; j < 4; ++j) {
            int n = n0 + j * 16 + l16;
            float bn_ = bias[n];
#pragma unroll
            for (int i = 0; i < 8; ++i) {
                int m = half * 8 + i;              // C/D: row = half*8 + vgpr
                Out[(size_t)(row0 + m) * NF + n]      = (f16)fmaxf(cr0[j][i] + bn_, 0.f);
                Out[(size_t)(row0 + 16 + m) * NF + n] = (f16)fmaxf(cr1[j][i] + bn_, 0.f);
            }
        }
    }
}

// ---------------------------------------------------------------------------
// Kernel 4: GEMM2 (512x512) for hero row r and villain row r+7680 in the same
// wave (shared B frags), fused bias+ReLU+difference+permuted store to d_out.
// diff row n = b*60+m  ->  output row m*128+b.
// ---------------------------------------------------------------------------
__global__ void __launch_bounds__(256, 1) gemm2(const f16* __restrict__ A,
                                                const f16* __restrict__ Bt,
                                                const float* __restrict__ bias,
                                                float* __restrict__ out) {
    const int lane = threadIdx.x & 31;
    const int wave = threadIdx.x >> 5;
    const int half = lane >> 4;
    const int l16  = lane & 15;
    const int tileM = blockIdx.x;                    // 0..479 (hero tiles)
    const int n0 = wave * 64;

    const f16* ah = A + (size_t)(tileM * 16 + l16) * NF;
    const f16* av = ah + (size_t)N_HERO * NF;
    const f16* bp = Bt + (size_t)(n0 + l16) * NF;

    v8f h0 = {}, h1 = {}, h2 = {}, h3 = {};
    v8f q0 = {}, q1 = {}, q2 = {}, q3 = {};
    for (int kk = 0; kk < NF; kk += 32) {
        v16h aa = load_frag(ah + kk, half);
        v16h az = load_frag(av + kk, half);
        v16h b0 = load_frag(bp + kk, half);
        v16h b1 = load_frag(bp + 16 * NF + kk, half);
        v16h b2 = load_frag(bp + 32 * NF + kk, half);
        v16h b3 = load_frag(bp + 48 * NF + kk, half);
        h0 = WMMA_F16(aa, b0, h0);  q0 = WMMA_F16(az, b0, q0);
        h1 = WMMA_F16(aa, b1, h1);  q1 = WMMA_F16(az, b1, q1);
        h2 = WMMA_F16(aa, b2, h2);  q2 = WMMA_F16(az, b2, q2);
        h3 = WMMA_F16(aa, b3, h3);  q3 = WMMA_F16(az, b3, q3);
    }
#pragma unroll
    for (int i = 0; i < 8; ++i) {
        int m = half * 8 + i;
        int rlog = tileM * 16 + m;                   // = b*60 + m_idx
        int bb = rlog / 60, mm = rlog - bb * 60;
        size_t orow = (size_t)(mm * 128 + bb) * NF;
        int n; float bn_;
        n = n0 + l16;      bn_ = bias[n];
        out[orow + n] = fmaxf(h0[i] + bn_, 0.f) - fmaxf(q0[i] + bn_, 0.f);
        n = n0 + 16 + l16; bn_ = bias[n];
        out[orow + n] = fmaxf(h1[i] + bn_, 0.f) - fmaxf(q1[i] + bn_, 0.f);
        n = n0 + 32 + l16; bn_ = bias[n];
        out[orow + n] = fmaxf(h2[i] + bn_, 0.f) - fmaxf(q2[i] + bn_, 0.f);
        n = n0 + 48 + l16; bn_ = bias[n];
        out[orow + n] = fmaxf(h3[i] + bn_, 0.f) - fmaxf(q3[i] + bn_, 0.f);
    }
}

// ---------------------------------------------------------------------------
extern "C" void kernel_launch(void* const* d_in, const int* in_sizes, int n_in,
                              void* d_out, int out_size, void* d_ws, size_t ws_size,
                              hipStream_t stream) {
    const int*   x    = (const int*)d_in[0];
    const float* ws_w = (const float*)d_in[1];
    const float* ws_b = (const float*)d_in[2];
    const float* wr_w = (const float*)d_in[3];
    const float* wr_b = (const float*)d_in[4];
    const float* sg   = (const float*)d_in[5];
    const float* sb   = (const float*)d_in[6];
    const float* sm   = (const float*)d_in[7];
    const float* sv   = (const float*)d_in[8];
    const float* rg   = (const float*)d_in[9];
    const float* rb   = (const float*)d_in[10];
    const float* rm   = (const float*)d_in[11];
    const float* rv   = (const float*)d_in[12];
    const float* w1   = (const float*)d_in[13];
    const float* b1   = (const float*)d_in[14];
    const float* w2   = (const float*)d_in[15];
    const float* b2   = (const float*)d_in[16];

    // Scratch layout:
    //   Wt1: 512*2048 f16 = 2 MiB     Wt2: 512*512 f16 = 512 KiB
    //   H:   15360*2048 f16 = 60 MiB  A1:  15360*512 f16 = 15 MiB
    char* ws = (char*)d_ws;
    f16* Wt1 = (f16*)(ws);
    f16* Wt2 = (f16*)(ws + (size_t)2 * 1024 * 1024);
    f16* H   = (f16*)(ws + (size_t)2 * 1024 * 1024 + 512 * 1024);
    f16* A1  = (f16*)(ws + (size_t)2 * 1024 * 1024 + 512 * 1024 +
                      (size_t)N_ROWS * K1 * sizeof(f16));

    prep_weights<<<(K1 * NF + NF * NF) / 256, 256, 0, stream>>>(w1, w2, Wt1, Wt2);
    features<<<(N_ROWS * 128) / 256, 256, 0, stream>>>(
        x, ws_w, ws_b, wr_w, wr_b, sg, sb, sm, sv, rg, rb, rm, rv, H);
    gemm1<<<N_ROWS / 32, 256, 0, stream>>>(H, Wt1, b1, A1);
    gemm2<<<N_HERO / 16, 256, 0, stream>>>(A1, Wt2, b2, (float*)d_out);
}
